// LSTM_Decoder_6347961663672
// MI455X (gfx1250) — matrix operand order repaired and checked by reading, not compiled, not run on hardware
//
#include <hip/hip_runtime.h>

// ---------------------------------------------------------------------------
// LSTM decoder for MI455X (gfx1250), wave32 + WMMA bf16, 2x2 register blocking.
// B=64, T=256, F=128, U=512 (latent), H1=300 (padded to 320), 4U=2048.
// ---------------------------------------------------------------------------

typedef __bf16 bf16;
typedef __attribute__((ext_vector_type(16))) __bf16 v16bf;
typedef __attribute__((ext_vector_type(8)))  float  v8f;

#define B_   64
#define T_   256
#define F_   128
#define U_   512
#define G_   2048      // 4*U
#define H_   300
#define HP_  320       // H1 padded to multiple of 32 (WMMA K depth)
#define BT_  (B_ * T_) // 16384 rows for the TimeDistributed dense chain

// ---------------- helpers ----------------

__device__ __forceinline__ bf16 f2bf(float f) {
    unsigned x = __builtin_bit_cast(unsigned, f);
    unsigned r = (x + 0x7FFFu + ((x >> 16) & 1u)) >> 16;  // round-to-nearest-even
    unsigned short h = (unsigned short)r;
    return __builtin_bit_cast(bf16, h);
}

__device__ __forceinline__ float sigmoidf(float x) {
    return 1.0f / (1.0f + __expf(-x));
}

union FragAB { unsigned u[8]; v16bf v; };

// A fragment: A row-major [M][lda] bf16, tile origin (m0, kb).
// ISA layout (16-bit A 16x32): lanes 0-15 M=lane, v0..3 K={0..7}, v4..7 K={16..23};
// lanes 16-31: same M, K shifted by +8 within each group.
__device__ __forceinline__ v16bf load_frag_a(const bf16* __restrict__ A, int lda,
                                             int m0, int kb, int lane) {
    const int half = (lane >> 4) & 1;
    const int mr   = lane & 15;
    const bf16* row = A + (size_t)(m0 + mr) * (size_t)lda;
    FragAB f;
#pragma unroll
    for (int r = 0; r < 8; ++r) {
        const int k = kb + ((r < 4) ? 0 : 16) + half * 8 + 2 * (r & 3);
        f.u[r] = *(const unsigned*)(row + k);
    }
    return f.v;
}

// B fragment from Bt stored N-major ([N][ldb], K contiguous), tile origin (n0, kb).
// ISA layout (32x16 B): lanes 0-15 v0..3 K={0..7}, v4..7 K={16..23}; lanes 16-31 K shifted by +8; N=lane%16.
__device__ __forceinline__ v16bf load_frag_b(const bf16* __restrict__ Bt, int ldb,
                                             int n0, int kb, int lane) {
    const int half = (lane >> 4) & 1;
    const int mr   = lane & 15;
    const bf16* row = Bt + (size_t)(n0 + mr) * (size_t)ldb;
    FragAB f;
#pragma unroll
    for (int r = 0; r < 8; ++r) {
        const int k = kb + ((r < 4) ? 0 : 16) + half * 8 + 2 * (r & 3);
        f.u[r] = *(const unsigned*)(row + k);
    }
    return f.v;
}

// ---------------- prep kernels ----------------

// out[n][k] = in[k][n] (f32 -> bf16), zero-padded outside [inRows x inCols].
__global__ void transpose_f32_to_bf16(const float* __restrict__ in, int inRows, int inCols,
                                      bf16* __restrict__ out, int outRows, int outCols) {
    int idx = blockIdx.x * blockDim.x + threadIdx.x;
    if (idx >= outRows * outCols) return;
    int n = idx / outCols;
    int k = idx - n * outCols;
    float v = (n < inCols && k < inRows) ? in[(size_t)k * inCols + n] : 0.0f;
    out[idx] = f2bf(v);
}

__global__ void convert_f32_to_bf16(const float* __restrict__ in, bf16* __restrict__ out, int n) {
    int idx = blockIdx.x * blockDim.x + threadIdx.x;
    if (idx < n) out[idx] = f2bf(in[idx]);
}

__global__ void pad_bias(const float* __restrict__ in, int n, float* __restrict__ out, int npad) {
    int idx = blockIdx.x * blockDim.x + threadIdx.x;
    if (idx < npad) out[idx] = (idx < n) ? in[idx] : 0.0f;
}

__global__ void zero_state(bf16* __restrict__ h0, float* __restrict__ c0, int n) {
    int idx = blockIdx.x * blockDim.x + threadIdx.x;
    if (idx < n) { h0[idx] = f2bf(0.0f); c0[idx] = 0.0f; }
}

// ---------------- generic WMMA GEMM, 2x2 tiles per wave ----------------
// C[M,N] = act(A[M,K] @ Bt[N,K]^T + bias). Each wave owns a 32x32 output block
// (2x2 WMMA tiles): A frags reused across 2 N-tiles, B frags across 2 M-tiles.
// Requires M,N multiples of 32. outF != nullptr -> f32 out, else bf16 to outB.
__global__ void gemm_bf16(const bf16* __restrict__ A, int lda,
                          const bf16* __restrict__ Bt, int ldb,
                          const float* __restrict__ bias,
                          float* __restrict__ outF,
                          bf16* __restrict__ outB,
                          int ldc, int Mblk, int Nblk, int K, int act) {
    const int wave = threadIdx.x >> 5;
    const int lane = threadIdx.x & 31;
    const int wid  = blockIdx.x * (blockDim.x >> 5) + wave;
    if (wid >= Mblk * Nblk) return;   // wave-uniform exit (EXEC stays all-ones)
    const int tm = wid / Nblk;
    const int tn = wid - tm * Nblk;
    const int m0 = tm * 32, n0 = tn * 32;

    v8f acc00 = {}, acc01 = {}, acc10 = {}, acc11 = {};
    for (int kb = 0; kb < K; kb += 32) {
        if (kb + 32 < K) {  // pull next K-block of the (hotter) B stream into WGP$
            __builtin_prefetch(Bt + (size_t)(n0 + (lane & 15)) * (size_t)ldb + kb + 32, 0, 1);
        }
        v16bf a0 = load_frag_a(A,  lda, m0,      kb, lane);
        v16bf a1 = load_frag_a(A,  lda, m0 + 16, kb, lane);
        v16bf b0 = load_frag_b(Bt, ldb, n0,      kb, lane);
        v16bf b1 = load_frag_b(Bt, ldb, n0 + 16, kb, lane);
        acc00 = __builtin_amdgcn_wmma_f32_16x16x32_bf16(false, a0, false, b0, (short)0, acc00, false, false);
        acc01 = __builtin_amdgcn_wmma_f32_16x16x32_bf16(false, a0, false, b1, (short)0, acc01, false, false);
        acc10 = __builtin_amdgcn_wmma_f32_16x16x32_bf16(false, a1, false, b0, (short)0, acc10, false, false);
        acc11 = __builtin_amdgcn_wmma_f32_16x16x32_bf16(false, a1, false, b1, (short)0, acc11, false, false);
    }

    const int half = (lane >> 4) & 1;
    const int mr   = lane & 15;
    const float bv0 = bias ? bias[n0 + mr]      : 0.0f;
    const float bv1 = bias ? bias[n0 + 16 + mr] : 0.0f;
#pragma unroll
    for (int j = 0; j < 8; ++j) {
        const int r0 = m0 + j + half * 8;
        const int r1 = r0 + 16;
        float v00 = acc00[j] + bv0, v01 = acc01[j] + bv1;
        float v10 = acc10[j] + bv0, v11 = acc11[j] + bv1;
        if (act == 1) {
            v00 = sigmoidf(v00); v01 = sigmoidf(v01);
            v10 = sigmoidf(v10); v11 = sigmoidf(v11);
        }
        if (outF) {
            outF[(size_t)r0 * ldc + n0 + mr]      = v00;
            outF[(size_t)r0 * ldc + n0 + 16 + mr] = v01;
            outF[(size_t)r1 * ldc + n0 + mr]      = v10;
            outF[(size_t)r1 * ldc + n0 + 16 + mr] = v11;
        } else {
            outB[(size_t)r0 * ldc + n0 + mr]      = f2bf(v00);
            outB[(size_t)r0 * ldc + n0 + 16 + mr] = f2bf(v01);
            outB[(size_t)r1 * ldc + n0 + mr]      = f2bf(v10);
            outB[(size_t)r1 * ldc + n0 + 16 + mr] = f2bf(v11);
        }
    }
}

// ---------------- LSTM recurrence step ----------------
// Block = 128 threads = 4 waves; wave g computes gate g's tiles for 32 batch
// rows (2 M-tiles sharing one B fragment of R) x one 16-unit column tile.
// Gate tiles exchanged via LDS, then c' = f*c + i*relu(cg), h' = o*relu(c').
__global__ void lstm_step(const float* __restrict__ xW,     // [64][2048] f32
                          const bf16*  __restrict__ Rt,     // [2048][512] bf16 (N-major)
                          const bf16*  __restrict__ h_prev, // [64][512] bf16
                          float*       __restrict__ c_state,// [64][512] f32 (in-place)
                          bf16*        __restrict__ h_next, // [64][512] bf16
                          bf16*        __restrict__ h_all,  // [B*T][512] bf16 (row = b*T + t)
                          int t) {
    __shared__ float gbuf[4][32][16];

    const int gate = threadIdx.x >> 5;    // 0:i 1:f 2:cg 3:o (Keras gate order)
    const int lane = threadIdx.x & 31;
    const int mp    = blockIdx.x >> 5;    // 0..1  (two 32-row halves of B=64)
    const int utile = blockIdx.x & 31;    // 0..31 (512 units)
    const int m0 = mp * 32;
    const int u0 = utile * 16;
    const int n0 = gate * U_ + u0;

    v8f acc0 = {}, acc1 = {};
    for (int kb = 0; kb < U_; kb += 32) {
        if (kb + 32 < U_) {
            __builtin_prefetch(Rt + (size_t)(n0 + (lane & 15)) * U_ + kb + 32, 0, 1);
        }
        v16bf a0 = load_frag_a(h_prev, U_, m0,      kb, lane);
        v16bf a1 = load_frag_a(h_prev, U_, m0 + 16, kb, lane);
        v16bf b  = load_frag_b(Rt,     U_, n0,      kb, lane);
        acc0 = __builtin_amdgcn_wmma_f32_16x16x32_bf16(false, a0, false, b, (short)0, acc0, false, false);
        acc1 = __builtin_amdgcn_wmma_f32_16x16x32_bf16(false, a1, false, b, (short)0, acc1, false, false);
    }

    const int half = (lane >> 4) & 1;
    const int mr   = lane & 15;
#pragma unroll
    for (int j = 0; j < 8; ++j) {
        const int row = j + half * 8;
        gbuf[gate][row][mr]      = acc0[j] + xW[(size_t)(m0 + row) * G_ + n0 + mr];
        gbuf[gate][row + 16][mr] = acc1[j] + xW[(size_t)(m0 + row + 16) * G_ + n0 + mr];
    }
    __syncthreads();

    // 128 threads cover the 512-element (32 batch x 16 unit) tile: 4 each.
#pragma unroll
    for (int e = threadIdx.x; e < 512; e += 128) {
        const int row = e >> 4, col = e & 15;
        const float gi = sigmoidf(gbuf[0][row][col]);
        const float gf = sigmoidf(gbuf[1][row][col]);
        const float gc = fmaxf(gbuf[2][row][col], 0.0f);   // activation = relu
        const float go = sigmoidf(gbuf[3][row][col]);
        const int br = m0 + row;
        const int un = u0 + col;
        const size_t idx = (size_t)br * U_ + un;
        const float cn = gf * c_state[idx] + gi * gc;
        c_state[idx] = cn;
        const bf16 hb = f2bf(go * fmaxf(cn, 0.0f));
        h_next[idx] = hb;
        h_all[((size_t)br * T_ + t) * U_ + un] = hb;
    }
}

// ---------------- launch ----------------

extern "C" void kernel_launch(void* const* d_in, const int* in_sizes, int n_in,
                              void* d_out, int out_size, void* d_ws, size_t ws_size,
                              hipStream_t stream) {
    const float* inputs = (const float*)d_in[0];
    const float* W  = (const float*)d_in[1];
    const float* R  = (const float*)d_in[2];
    const float* b  = (const float*)d_in[3];
    const float* W1 = (const float*)d_in[4];
    const float* b1 = (const float*)d_in[5];
    const float* W2 = (const float*)d_in[6];
    const float* b2 = (const float*)d_in[7];
    const float* W3 = (const float*)d_in[8];
    const float* b3 = (const float*)d_in[9];
    float* out = (float*)d_out;

    char* base = (char*)d_ws;
    size_t off = 0;
    auto alloc = [&](size_t bytes) -> void* {
        void* r = base + off;
        off = (off + bytes + 255) & ~(size_t)255;
        return r;
    };
    bf16*  Rt    = (bf16*)alloc((size_t)G_ * U_ * sizeof(bf16));   // R^T  [2048][512]
    bf16*  Wt    = (bf16*)alloc((size_t)G_ * U_ * sizeof(bf16));   // W^T  [2048][512]
    bf16*  W1t   = (bf16*)alloc((size_t)HP_ * U_ * sizeof(bf16));  // W1^T [320][512]
    bf16*  W2t   = (bf16*)alloc((size_t)HP_ * HP_ * sizeof(bf16)); // W2^T [320][320]
    bf16*  W3t   = (bf16*)alloc((size_t)F_ * HP_ * sizeof(bf16));  // W3^T [128][320]
    bf16*  inb   = (bf16*)alloc((size_t)B_ * U_ * sizeof(bf16));   // inputs bf16
    float* b1p   = (float*)alloc(HP_ * sizeof(float));
    float* b2p   = (float*)alloc(HP_ * sizeof(float));
    float* xW    = (float*)alloc((size_t)B_ * G_ * sizeof(float)); // [64][2048]
    bf16*  hA    = (bf16*)alloc((size_t)B_ * U_ * sizeof(bf16));
    bf16*  hB    = (bf16*)alloc((size_t)B_ * U_ * sizeof(bf16));
    float* cSt   = (float*)alloc((size_t)B_ * U_ * sizeof(float));
    bf16*  h_all = (bf16*)alloc((size_t)BT_ * U_ * sizeof(bf16));  // 16 MB
    bf16*  x1    = (bf16*)alloc((size_t)BT_ * HP_ * sizeof(bf16)); // 10 MB
    bf16*  x2    = (bf16*)alloc((size_t)BT_ * HP_ * sizeof(bf16)); // 10 MB
    (void)ws_size; (void)in_sizes; (void)n_in; (void)out_size;

    const int THR = 256;  // 8 waves/block
    auto blk = [](int n, int t) { return (n + t - 1) / t; };

    // --- weight prep (bf16, transposed/padded) ---
    transpose_f32_to_bf16<<<blk(G_ * U_, THR), THR, 0, stream>>>(W,  U_, G_, Wt,  G_,  U_);
    transpose_f32_to_bf16<<<blk(G_ * U_, THR), THR, 0, stream>>>(R,  U_, G_, Rt,  G_,  U_);
    transpose_f32_to_bf16<<<blk(HP_ * U_, THR), THR, 0, stream>>>(W1, U_, H_, W1t, HP_, U_);
    transpose_f32_to_bf16<<<blk(HP_ * HP_, THR), THR, 0, stream>>>(W2, H_, H_, W2t, HP_, HP_);
    transpose_f32_to_bf16<<<blk(F_ * HP_, THR), THR, 0, stream>>>(W3, H_, F_, W3t, F_,  HP_);
    convert_f32_to_bf16<<<blk(B_ * U_, THR), THR, 0, stream>>>(inputs, inb, B_ * U_);
    pad_bias<<<blk(HP_, THR), THR, 0, stream>>>(b1, H_, b1p, HP_);
    pad_bias<<<blk(HP_, THR), THR, 0, stream>>>(b2, H_, b2p, HP_);
    zero_state<<<blk(B_ * U_, THR), THR, 0, stream>>>(hA, cSt, B_ * U_);

    // --- xW = inputs @ W + b  (f32 out, reused every step) ---
    // 32x32 blocks: Mblk=2, Nblk=64 -> 128 wave-jobs / 8 = 16 blocks
    gemm_bf16<<<(B_ / 32) * (G_ / 32) / 8, THR, 0, stream>>>(
        inb, U_, Wt, U_, b, xW, nullptr, G_, B_ / 32, G_ / 32, U_, 0);

    // --- LSTM recurrence: 256 sequential launches (stream = global sync) ---
    for (int t = 0; t < T_; ++t) {
        const bf16* hp = (t & 1) ? hB : hA;
        bf16*       hn = (t & 1) ? hA : hB;
        lstm_step<<<64, 128, 0, stream>>>(xW, Rt, hp, cSt, hn, h_all, t);
    }

    // --- TimeDistributed dense chain over 16384 rows ---
    // x1 = h_all @ W1 + b1   (bf16 out)  Mblk=512, Nblk=10 -> 5120/8 = 640 blocks
    gemm_bf16<<<(BT_ / 32) * (HP_ / 32) / 8, THR, 0, stream>>>(
        h_all, U_, W1t, U_, b1p, nullptr, x1, HP_, BT_ / 32, HP_ / 32, U_, 0);
    // x2 = sigmoid(x1 @ W2 + b2)   (bf16 out)
    gemm_bf16<<<(BT_ / 32) * (HP_ / 32) / 8, THR, 0, stream>>>(
        x1, HP_, W2t, HP_, b2p, nullptr, x2, HP_, BT_ / 32, HP_ / 32, HP_, 1);
    // out = x2 @ W3 + b3   (f32 out)  Mblk=512, Nblk=4 -> 2048/8 = 256 blocks
    gemm_bf16<<<(BT_ / 32) * (F_ / 32) / 8, THR, 0, stream>>>(
        x2, HP_, W3t, HP_, b3, out, nullptr, F_, BT_ / 32, F_ / 32, HP_, 0);
}
